// TriangleMultiplicativeUpdate_88948772700514
// MI455X (gfx1250) — compile-verified, matrix operand order checked by reading, not run on hardware
//
#include <hip/hip_runtime.h>
#include <stdint.h>

#define NSEQ 512
#define DCH  128
#define NN   (NSEQ * NSEQ)   // 262144 rows
#define EPS  1e-5f

typedef __attribute__((ext_vector_type(16))) __bf16 v16bf;
typedef __attribute__((ext_vector_type(8)))  float  v8f;

union Frag { uint4 u[2]; v16bf v; };

__device__ __forceinline__ unsigned int f2bf(float f) {
    unsigned int u = __float_as_uint(f);
    return ((u + 0x7FFFu + ((u >> 16) & 1u)) >> 16) & 0xFFFFu;   // RNE f32->bf16
}
__device__ __forceinline__ float bf2f(unsigned int us) {
    return __uint_as_float(us << 16);
}
__device__ __forceinline__ float sigm(float x) { return 1.0f / (1.0f + __expf(-x)); }

// ---------------------------------------------------------------------------
// k0: pack six 128x128 f32 weights into WMMA-B bf16 fragment order.
// Fragment index (w*32 + nt*4 + kt); within: [lane][8 dword pairs].
// Pair p of lane L holds W[k][n],W[k+1][n] with n = nt*16 + (L&15),
// k = kt*32 + (L>>4)*8 + (p<4 ? 2p : 16 + 2(p-4))   (ISA 16-bit B layout).
// ---------------------------------------------------------------------------
__global__ void __launch_bounds__(256) k0_pack(
    const float* __restrict__ Wga, const float* __restrict__ Wa,
    const float* __restrict__ Wgb, const float* __restrict__ Wb,
    const float* __restrict__ Wgo, const float* __restrict__ Wpo,
    unsigned int* __restrict__ wpack)
{
    const int idx = blockIdx.x * 256 + threadIdx.x;
    if (idx >= 6 * 8192) return;
    const int w    = idx >> 13;
    const int rem  = idx & 8191;
    const int nt   = rem >> 10;
    const int kt   = (rem >> 8) & 3;
    const int lane = (rem >> 3) & 31;
    const int p    = rem & 7;
    const float* W = (w == 0) ? Wga : (w == 1) ? Wa : (w == 2) ? Wgb
                   : (w == 3) ? Wb  : (w == 4) ? Wgo : Wpo;
    const int n    = nt * 16 + (lane & 15);
    const int half = lane >> 4;
    const int k    = kt * 32 + half * 8 + ((p < 4) ? p * 2 : 16 + (p - 4) * 2);
    wpack[idx] = f2bf(W[k * DCH + n]) | (f2bf(W[(k + 1) * DCH + n]) << 16);
}

// ---------------------------------------------------------------------------
// k1: fused input LayerNorm + 5 projections + gating/masking.
// 8 waves/block, each wave = one 16-row tile of the flattened (N*N, D) input.
// Per-nt, the 20 needed weight fragments (20KB) are staged into LDS once per
// block (8x fewer L2 reads than per-wave loads); staging is two-phase
// (clause of loads, then stores) so it pipelines. Outputs aT/bT/gT in
// d-major bf16 layout: t[d*NN + row].
// ---------------------------------------------------------------------------
__global__ void __launch_bounds__(256) k1_ln_proj(
    const float* __restrict__ z, const unsigned char* __restrict__ key_mask,
    const float* __restrict__ ln_g, const float* __restrict__ ln_b,
    const float* __restrict__ ba,  const float* __restrict__ bga,
    const float* __restrict__ bb,  const float* __restrict__ bgb,
    const float* __restrict__ bgo,
    const unsigned int* __restrict__ wpack,
    unsigned short* __restrict__ aT, unsigned short* __restrict__ bT,
    unsigned short* __restrict__ gT)
{
    __shared__ __align__(16) unsigned int lds1[8][16][68];   // zn, 4-bank rotate/row
    __shared__ __align__(16) unsigned int wlds[5 * 4 * 256]; // 20KB weight frags
    const int wave = threadIdx.x >> 5, lane = threadIdx.x & 31;
    const int rowbase = blockIdx.x * 128 + wave * 16;

    // ---- Phase 1: LN of 16 rows; 2 lanes per row, 64 columns each ----
    {
        const int rl   = lane >> 1;
        const int colb = (lane & 1) * 64;
        const float* zr = z + (size_t)(rowbase + rl) * DCH + colb;
        float4 xv[16];
        float s1 = 0.f, s2 = 0.f;
#pragma unroll
        for (int j = 0; j < 16; ++j) {
            xv[j] = *(const float4*)(zr + j * 4);
            s1 += xv[j].x + xv[j].y + xv[j].z + xv[j].w;
            s2 += xv[j].x * xv[j].x + xv[j].y * xv[j].y
                + xv[j].z * xv[j].z + xv[j].w * xv[j].w;
        }
        s1 += __shfl_xor(s1, 1, 32);
        s2 += __shfl_xor(s2, 1, 32);
        const float mean = s1 * (1.f / 128.f);
        const float var  = s2 * (1.f / 128.f) - mean * mean;
        const float rstd = rsqrtf(var + EPS);
#pragma unroll
        for (int j = 0; j < 16; ++j) {
            const float4 g4 = *(const float4*)(ln_g + colb + j * 4);
            const float4 b4 = *(const float4*)(ln_b + colb + j * 4);
            const float y0 = (xv[j].x - mean) * rstd * g4.x + b4.x;
            const float y1 = (xv[j].y - mean) * rstd * g4.y + b4.y;
            const float y2 = (xv[j].z - mean) * rstd * g4.z + b4.z;
            const float y3 = (xv[j].w - mean) * rstd * g4.w + b4.w;
            lds1[wave][rl][(colb >> 1) + j * 2]     = f2bf(y0) | (f2bf(y1) << 16);
            lds1[wave][rl][(colb >> 1) + j * 2 + 1] = f2bf(y2) | (f2bf(y3) << 16);
        }
    }
    __syncthreads();

    // ---- Phase 2: A fragments from LDS (ISA 16-bit A layout) ----
    const int m = lane & 15, half = lane >> 4;
    Frag afr[4];
#pragma unroll
    for (int kt = 0; kt < 4; ++kt) {
        const unsigned int* p = &lds1[wave][m][kt * 16 + half * 4];
        afr[kt].u[0] = *(const uint4*)p;         // K = kt*32 + half*8 .. +7
        afr[kt].u[1] = *(const uint4*)(p + 8);   // K = kt*32 + 16 + half*8 .. +7
    }
    const int kb = (rowbase & (NSEQ - 1)) + half * 8;
    unsigned int mbits = 0;
#pragma unroll
    for (int r = 0; r < 8; ++r) mbits |= key_mask[kb + r] ? (1u << r) : 0u;

#pragma unroll 1
    for (int nt = 0; nt < 8; ++nt) {
        // stage the 20 fragments for this nt: 1280 uint4, 5 per thread.
        // two-phase (loads first, stores second) so the loads issue as a clause
        uint4 st[5];
#pragma unroll
        for (int q = 0; q < 5; ++q) {
            const int idx  = q * 256 + threadIdx.x;       // uint4 index
            const int w    = idx >> 8;
            const int kt   = (idx >> 6) & 3;
            const int rest = idx & 63;
            st[q] = ((const uint4*)wpack)[((w * 32 + nt * 4 + kt) << 6) + rest];
        }
        __syncthreads();   // previous iteration's wlds reads complete
#pragma unroll
        for (int q = 0; q < 5; ++q)
            ((uint4*)wlds)[q * 256 + threadIdx.x] = st[q];
        __syncthreads();

        const int d = nt * 16 + m;
        v8f acc[5];
#pragma unroll
        for (int w = 0; w < 5; ++w) acc[w] = (v8f)0.0f;
#pragma unroll
        for (int w = 0; w < 5; ++w) {            // 0:Wga 1:Wa 2:Wgb 3:Wb 4:Wgo
#pragma unroll
            for (int kt = 0; kt < 4; ++kt) {
                Frag bfr;
                const unsigned int* q = wlds + ((w * 4 + kt) << 8) + lane * 8;
                bfr.u[0] = *(const uint4*)q;
                bfr.u[1] = *(const uint4*)(q + 4);
                acc[w] = __builtin_amdgcn_wmma_f32_16x16x32_bf16(
                    false, afr[kt].v, false, bfr.v, (short)0, acc[w], false, false);
            }
        }
        const float bga_d = bga[d], ba_d = ba[d];
        const float bgb_d = bgb[d], bb_d = bb[d], bgo_d = bgo[d];
        union { uint4 q; unsigned short s[8]; } pa, pb, pg;
#pragma unroll
        for (int r = 0; r < 8; ++r) {
            float va = sigm(acc[0][r] + bga_d) * (acc[1][r] + ba_d);
            float vb = sigm(acc[2][r] + bgb_d) * (acc[3][r] + bb_d);
            const float vg = sigm(acc[4][r] + bgo_d);
            if ((mbits >> r) & 1) { va = 0.f; vb = 0.f; }
            pa.s[r] = (unsigned short)f2bf(va);
            pb.s[r] = (unsigned short)f2bf(vb);
            pg.s[r] = (unsigned short)f2bf(vg);
        }
        // lane's 8 C-rows are consecutive -> one contiguous 16B store each
        const size_t o = (size_t)d * NN + rowbase + half * 8;
        *(uint4*)(aT + o) = pa.q;
        *(uint4*)(bT + o) = pb.q;
        *(uint4*)(gT + o) = pg.q;
    }
}

// ---------------------------------------------------------------------------
// k2: triangle einsum x[i,j,d] = sum_k a[i,k,d]*b[j,k,d].
// 128 independent 512x512 (K=512) GEMMs; aT/bT K-contiguous for both operands
// (NT layout), L2-resident (128MB < 192MB L2). Each wave: 32x32 output tile
// (2 A-frags x 2 B-frags -> 4 WMMAs/k-step, ~16 flops per L2 byte), with
// global_prefetch hints for the next k-step. 8 waves/block = 8 d-channels on
// one spatial tile; d-transpose through LDS so x is written d-fastest (32B).
// ---------------------------------------------------------------------------
__global__ void __launch_bounds__(256) k2_einsum(
    const unsigned short* __restrict__ aT, const unsigned short* __restrict__ bT,
    float* __restrict__ x)
{
    __shared__ float lx[8][32][32];   // 32KB
    const int wave = threadIdx.x >> 5, lane = threadIdx.x & 31;
    const int it = blockIdx.x & 15, jt = blockIdx.x >> 4;
    const int d  = blockIdx.y * 8 + wave;
    const int m = lane & 15, half = lane >> 4;
    const unsigned short* ap = aT + (size_t)d * NN + (size_t)(it * 32 + m) * NSEQ;
    const unsigned short* bp = bT + (size_t)d * NN + (size_t)(jt * 32 + m) * NSEQ;
    v8f acc00 = (v8f)0.0f, acc01 = (v8f)0.0f, acc10 = (v8f)0.0f, acc11 = (v8f)0.0f;
#pragma unroll 2
    for (int ks = 0; ks < 16; ++ks) {            // K = 512 = 16 x 32
        const int k0 = ks * 32 + half * 8;
        Frag a0, a1, b0, b1;
        a0.u[0] = *(const uint4*)(ap + k0);
        a0.u[1] = *(const uint4*)(ap + k0 + 16);
        a1.u[0] = *(const uint4*)(ap + 16 * NSEQ + k0);
        a1.u[1] = *(const uint4*)(ap + 16 * NSEQ + k0 + 16);
        b0.u[0] = *(const uint4*)(bp + k0);
        b0.u[1] = *(const uint4*)(bp + k0 + 16);
        b1.u[0] = *(const uint4*)(bp + 16 * NSEQ + k0);
        b1.u[1] = *(const uint4*)(bp + 16 * NSEQ + k0 + 16);
        if (ks < 15) {   // hint next k-step (lowers to global_prefetch_b8)
            __builtin_prefetch(ap + k0 + 32, 0, 1);
            __builtin_prefetch(bp + k0 + 32, 0, 1);
        }
        acc00 = __builtin_amdgcn_wmma_f32_16x16x32_bf16(
            false, a0.v, false, b0.v, (short)0, acc00, false, false);
        acc01 = __builtin_amdgcn_wmma_f32_16x16x32_bf16(
            false, a0.v, false, b1.v, (short)0, acc01, false, false);
        acc10 = __builtin_amdgcn_wmma_f32_16x16x32_bf16(
            false, a1.v, false, b0.v, (short)0, acc10, false, false);
        acc11 = __builtin_amdgcn_wmma_f32_16x16x32_bf16(
            false, a1.v, false, b1.v, (short)0, acc11, false, false);
    }
#pragma unroll
    for (int r = 0; r < 8; ++r) {
        lx[wave][r + half * 8][m]           = acc00[r];
        lx[wave][r + half * 8][m + 16]      = acc01[r];
        lx[wave][r + half * 8 + 16][m]      = acc10[r];
        lx[wave][r + half * 8 + 16][m + 16] = acc11[r];
    }
    __syncthreads();
    const int dbase = blockIdx.y * 8;
#pragma unroll
    for (int q = 0; q < 4; ++q) {   // 1024 (i,j) pairs, 4 per thread, 32B runs
        const int pi = q * 256 + threadIdx.x;
        const int il = pi >> 5, jl = pi & 31;
        const float4 o0 = make_float4(lx[0][il][jl], lx[1][il][jl],
                                      lx[2][il][jl], lx[3][il][jl]);
        const float4 o1 = make_float4(lx[4][il][jl], lx[5][il][jl],
                                      lx[6][il][jl], lx[7][il][jl]);
        float* xp = x + ((size_t)(it * 32 + il) * NSEQ + (jt * 32 + jl)) * DCH + dbase;
        *(float4*)xp       = o0;
        *(float4*)(xp + 4) = o1;
    }
}

// ---------------------------------------------------------------------------
// k3: fused output LayerNorm + (x @ Wpo + bpo) * sigmoid-gate. In-place on
// d_out: each wave reads its 16 rows fully before writing them back.
// Gate gT is d-major, so it loads exactly onto C-fragment lanes (16B/lane).
// Wpo fragments staged per-nt through LDS (once per block, not per wave).
// ---------------------------------------------------------------------------
__global__ void __launch_bounds__(256) k3_out(
    const float* __restrict__ xin,
    const float* __restrict__ ln_g, const float* __restrict__ ln_b,
    const unsigned int* __restrict__ wpack,
    const unsigned short* __restrict__ gT,
    const float* __restrict__ bpo,
    float* __restrict__ out)
{
    __shared__ __align__(16) unsigned int lds1[8][16][68];
    __shared__ __align__(16) unsigned int wlds[4 * 256];   // 4KB (4 kt frags)
    const int wave = threadIdx.x >> 5, lane = threadIdx.x & 31;
    const int rowbase = blockIdx.x * 128 + wave * 16;
    {
        const int rl   = lane >> 1;
        const int colb = (lane & 1) * 64;
        const float* xr = xin + (size_t)(rowbase + rl) * DCH + colb;
        float4 xv[16];
        float s1 = 0.f, s2 = 0.f;
#pragma unroll
        for (int j = 0; j < 16; ++j) {
            xv[j] = *(const float4*)(xr + j * 4);
            s1 += xv[j].x + xv[j].y + xv[j].z + xv[j].w;
            s2 += xv[j].x * xv[j].x + xv[j].y * xv[j].y
                + xv[j].z * xv[j].z + xv[j].w * xv[j].w;
        }
        s1 += __shfl_xor(s1, 1, 32);
        s2 += __shfl_xor(s2, 1, 32);
        const float mean = s1 * (1.f / 128.f);
        const float var  = s2 * (1.f / 128.f) - mean * mean;
        const float rstd = rsqrtf(var + EPS);
#pragma unroll
        for (int j = 0; j < 16; ++j) {
            const float4 g4 = *(const float4*)(ln_g + colb + j * 4);
            const float4 b4 = *(const float4*)(ln_b + colb + j * 4);
            const float y0 = (xv[j].x - mean) * rstd * g4.x + b4.x;
            const float y1 = (xv[j].y - mean) * rstd * g4.y + b4.y;
            const float y2 = (xv[j].z - mean) * rstd * g4.z + b4.z;
            const float y3 = (xv[j].w - mean) * rstd * g4.w + b4.w;
            lds1[wave][rl][(colb >> 1) + j * 2]     = f2bf(y0) | (f2bf(y1) << 16);
            lds1[wave][rl][(colb >> 1) + j * 2 + 1] = f2bf(y2) | (f2bf(y3) << 16);
        }
    }
    __syncthreads();

    const int m = lane & 15, half = lane >> 4;
    Frag afr[4];
#pragma unroll
    for (int kt = 0; kt < 4; ++kt) {
        const unsigned int* p = &lds1[wave][m][kt * 16 + half * 4];
        afr[kt].u[0] = *(const uint4*)p;
        afr[kt].u[1] = *(const uint4*)(p + 8);
    }
#pragma unroll 1
    for (int nt = 0; nt < 8; ++nt) {
        // stage Wpo fragments for this nt: 256 uint4, 1 per thread (two-phase)
        const int kt0  = threadIdx.x >> 6;
        const int rest = threadIdx.x & 63;
        const uint4 st =
            ((const uint4*)wpack)[((5 * 32 + nt * 4 + kt0) << 6) + rest];
        __syncthreads();
        ((uint4*)wlds)[threadIdx.x] = st;
        __syncthreads();

        const int d = nt * 16 + m;
        v8f acc = (v8f)0.0f;
#pragma unroll
        for (int kt = 0; kt < 4; ++kt) {
            Frag bfr;
            const unsigned int* q = wlds + (kt << 8) + lane * 8;
            bfr.u[0] = *(const uint4*)q;
            bfr.u[1] = *(const uint4*)(q + 4);
            acc = __builtin_amdgcn_wmma_f32_16x16x32_bf16(
                false, afr[kt].v, false, bfr.v, (short)0, acc, false, false);
        }
        const float bpo_d = bpo[d];
        union { uint4 q; unsigned short s[8]; } gv;
        gv.q = *(const uint4*)(gT + (size_t)d * NN + rowbase + half * 8);
#pragma unroll
        for (int r = 0; r < 8; ++r) {
            out[(size_t)(rowbase + r + half * 8) * DCH + d]
                = bf2f(gv.s[r]) * (acc[r] + bpo_d);
        }
    }
}

// ---------------------------------------------------------------------------
extern "C" void kernel_launch(void* const* d_in, const int* in_sizes, int n_in,
                              void* d_out, int out_size, void* d_ws, size_t ws_size,
                              hipStream_t stream)
{
    (void)in_sizes; (void)n_in; (void)out_size; (void)ws_size;
    const float* z              = (const float*)d_in[0];
    const unsigned char* kmask  = (const unsigned char*)d_in[1];
    const float* ln_in_g = (const float*)d_in[2];
    const float* ln_in_b = (const float*)d_in[3];
    const float* Wa  = (const float*)d_in[4];
    const float* ba  = (const float*)d_in[5];
    const float* Wga = (const float*)d_in[6];
    const float* bga = (const float*)d_in[7];
    const float* Wb  = (const float*)d_in[8];
    const float* bb  = (const float*)d_in[9];
    const float* Wgb = (const float*)d_in[10];
    const float* bgb = (const float*)d_in[11];
    const float* Wgo = (const float*)d_in[12];
    const float* bgo = (const float*)d_in[13];
    const float* Wpo = (const float*)d_in[14];
    const float* bpo = (const float*)d_in[15];
    const float* ln_out_g = (const float*)d_in[16];
    const float* ln_out_b = (const float*)d_in[17];

    char* ws = (char*)d_ws;
    unsigned short* aT = (unsigned short*)(ws);                               // 64 MB
    unsigned short* bT = (unsigned short*)(ws + (size_t)64 * 1024 * 1024);    // 64 MB
    unsigned short* gT = (unsigned short*)(ws + (size_t)128 * 1024 * 1024);   // 64 MB
    unsigned int* wpack = (unsigned int*)(ws + (size_t)192 * 1024 * 1024);    // 192 KB
    float* xbuf = (float*)d_out;   // einsum result staged in d_out; k3 rewrites in place

    k0_pack<<<192, 256, 0, stream>>>(Wga, Wa, Wgb, Wb, Wgo, Wpo, wpack);
    k1_ln_proj<<<2048, 256, 0, stream>>>(z, kmask, ln_in_g, ln_in_b,
                                         ba, bga, bb, bgb, bgo, wpack, aT, bT, gT);
    k2_einsum<<<dim3(256, 16), 256, 0, stream>>>(aT, bT, xbuf);
    k3_out<<<2048, 256, 0, stream>>>(xbuf, ln_out_g, ln_out_b, wpack, gT, bpo,
                                     (float*)d_out);
}
